// Attention_1898375545286
// MI455X (gfx1250) — matrix operand order repaired and checked by reading, not compiled, not run on hardware
//
#include <hip/hip_runtime.h>
#include <hip/hip_bf16.h>

#define NRES 256   // N (and Q/K length)
#define CIN  128   // input channels
#define NH   4     // heads
#define CH   32    // per-head channels
#define QBLK 64    // query rows per attention workgroup

typedef __attribute__((ext_vector_type(16))) __bf16 v16bf;
typedef __attribute__((ext_vector_type(8)))  __bf16 v8bf;
typedef __attribute__((ext_vector_type(8)))  float  v8f;

static __device__ __forceinline__ v8f wmma_bf16(v16bf a, v16bf b, v8f c) {
  // D(16x16 f32) = A(16x32 bf16) * B(32x16 bf16) + C
  return __builtin_amdgcn_wmma_f32_16x16x32_bf16(false, a, false, b, (short)0, c,
                                                 false, false);
}

// Fast reciprocal (v_rcp_f32): fine for sigmoid gates / softmax normalizer.
static __device__ __forceinline__ float fast_rcp(float x) {
  return __builtin_amdgcn_rcpf(x);
}

// A fragment: lane half h -> K = 16*h + e (contiguous 16-element run from
// (row, ktile*32 + half*16)), matching the B fragments. Vector-load it.
static __device__ __forceinline__ v16bf load_A(const __bf16* p /* row base + half*16 */) {
  return *(const v16bf*)(p);
}

// ---------------------------------------------------------------------------
// Tensor Data Mover: 1-D tile load (contiguous bf16 block) into LDS.
// D# packed per CDNA5 ISA. This toolchain declares the
// 6-arg builtin: (u32x4 g0, i32x8 g1, i32x4 g2, i32x4 g3, i32x8, i32 cpol).
// ---------------------------------------------------------------------------
#if defined(__gfx1250__) && __has_builtin(__builtin_amdgcn_tensor_load_to_lds) && \
    __has_builtin(__builtin_amdgcn_s_wait_tensorcnt)
#define USE_TDM 1
typedef __attribute__((ext_vector_type(4))) unsigned int u32x4;
typedef __attribute__((ext_vector_type(8))) int i32x8;
typedef __attribute__((ext_vector_type(4))) int i32x4;

static __device__ __forceinline__ void tdm_load_1d(const void* gptr, void* lds,
                                                   unsigned nelem /* bf16 count */) {
  const unsigned long long ga = (unsigned long long)(uintptr_t)gptr;
  u32x4 g0;
  g0[0] = 1u;                                   // count=1, user mode, no gather
  g0[1] = (unsigned)(uintptr_t)lds;             // lds_addr (low 32 bits = LDS offset)
  g0[2] = (unsigned)(ga & 0xffffffffu);         // global_addr[31:0]
  g0[3] = (unsigned)((ga >> 32) & 0x1ffffffu)   // global_addr[56:32]
          | (2u << 30);                         // type = 2 ("image")
  i32x8 g1;
  g1[0] = (int)(1u << 16);                      // workgroup_mask=0, data_size=1 (2B)
  g1[1] = (int)((nelem & 0xffffu) << 16);       // tensor_dim0[15:0]
  g1[2] = (int)((nelem >> 16) | (1u << 16));    // tensor_dim0[31:16] | tensor_dim1=1
  g1[3] = (int)((nelem & 0xffffu) << 16);       // tile_dim0 = nelem
  g1[4] = 1;                                    // tile_dim1=1, tile_dim2=0
  g1[5] = (int)nelem;                           // tensor_dim0_stride[31:0]
  g1[6] = (int)((nelem & 0xffffu) << 16);       // stride0[47:32]=0 | stride1[15:0]
  g1[7] = 0;                                    // tensor_dim1_stride[47:16]
  __builtin_amdgcn_tensor_load_to_lds(g0, g1, i32x4{}, i32x4{}, i32x8{}, 0);
}
#endif

// ---------------------------------------------------------------------------
// Kernel 1: projections. q = (q_x@Wq)/sqrt(CH), k = kv_x@Wk, v = kv_x@Wv
// (bf16, layout [n][h][row][c]) and g = sigmoid(q_x@Wg) (fp32).
// 128 output rows per workgroup; weights staged transposed in LDS, two per
// pass (128 KB LDS total).
// ---------------------------------------------------------------------------
__global__ __launch_bounds__(256) void tri_proj_kernel(
    const float* __restrict__ qx, const float* __restrict__ kvx,
    const float* __restrict__ Wq, const float* __restrict__ Wk,
    const float* __restrict__ Wv, const float* __restrict__ Wg,
    __bf16* __restrict__ qo, __bf16* __restrict__ ko, __bf16* __restrict__ vo,
    float* __restrict__ go)
{
  __shared__ __bf16 sXq[128 * CIN];        // 32 KB
  __shared__ __bf16 sXk[128 * CIN];        // 32 KB
  __shared__ __bf16 sWt[2][CIN * CIN];     // 64 KB, transposed [n][k], 2 per pass

  const int tid  = threadIdx.x;
  const int lane = tid & 31;
  const int wave = tid >> 5;      // 0..7
  const int half = lane >> 4;     // 0/1
  const int lm   = lane & 15;
  const size_t m_base = (size_t)blockIdx.x * 128;

  const float* Ws[4] = {Wq, Wk, Wv, Wg};
  const float QSCALE = 0.17677669529663688f;   // 1/sqrt(32)

  // Stage activations (f32 -> bf16), vectorized.
  for (int i = tid * 4; i < 128 * CIN; i += 1024) {
    float4 a = *(const float4*)&qx[m_base * CIN + i];
    float4 b = *(const float4*)&kvx[m_base * CIN + i];
    sXq[i + 0] = (__bf16)a.x; sXq[i + 1] = (__bf16)a.y;
    sXq[i + 2] = (__bf16)a.z; sXq[i + 3] = (__bf16)a.w;
    sXk[i + 0] = (__bf16)b.x; sXk[i + 1] = (__bf16)b.y;
    sXk[i + 2] = (__bf16)b.z; sXk[i + 3] = (__bf16)b.w;
  }

  v16bf aq[4], ak[4];

#pragma unroll 1
  for (int pass = 0; pass < 2; ++pass) {
    if (pass) __syncthreads();   // previous pass reads complete before restage
    // Stage two weight matrices, transposed -> sWt[wi][n*CIN + k]
    for (int wi = 0; wi < 2; ++wi) {
      const float* W = Ws[pass * 2 + wi];
      for (int i = tid * 4; i < CIN * CIN; i += 1024) {
        const int k = i >> 7, n = i & 127;
        float4 wv = *(const float4*)&W[i];
        sWt[wi][(n + 0) * CIN + k] = (__bf16)wv.x;
        sWt[wi][(n + 1) * CIN + k] = (__bf16)wv.y;
        sWt[wi][(n + 2) * CIN + k] = (__bf16)wv.z;
        sWt[wi][(n + 3) * CIN + k] = (__bf16)wv.w;
      }
    }
    __syncthreads();

    const int m0 = wave * 16;
    if (pass == 0) {
#pragma unroll
      for (int kt = 0; kt < 4; ++kt) {
        aq[kt] = load_A(&sXq[(m0 + lm) * CIN + kt * 32 + (half << 4)]);
        ak[kt] = load_A(&sXk[(m0 + lm) * CIN + kt * 32 + (half << 4)]);
      }
    }

#pragma unroll 1
    for (int wi = 0; wi < 2; ++wi) {
      const int w = pass * 2 + wi;
      const v16bf* A = (w == 0 || w == 3) ? &aq[0] : &ak[0];
#pragma unroll 1
      for (int nt = 0; nt < 8; ++nt) {
        const int n0 = nt * 16;
        v8f acc = {};
#pragma unroll
        for (int kt = 0; kt < 4; ++kt) {
          v16bf b = *(const v16bf*)&sWt[wi][(n0 + lm) * CIN + kt * 32 + (half << 4)];
          acc = wmma_bf16(A[kt], b, acc);
        }
#pragma unroll
        for (int r = 0; r < 8; ++r) {
          const size_t M = m_base + m0 + r + 8 * half;   // global row 0..65535
          const int j = n0 + lm;                         // output col 0..127
          if (w == 3) {
            go[M * CIN + j] = fast_rcp(1.0f + __expf(-acc[r]));
          } else {
            const int h = j >> 5, c = j & 31;
            const size_t nres = M >> 8;
            const size_t rrow = M & 255;
            const size_t idx = (((nres * NH + h) * NRES) + rrow) * CH + c;
            float val = acc[r];
            if (w == 0) val *= QSCALE;
            __bf16* dst = (w == 0) ? qo : (w == 1 ? ko : vo);
            dst[idx] = (__bf16)val;
          }
        }
      }
    }
  }
}

// ---------------------------------------------------------------------------
// Kernel 2: attention for one (n, h, 64-row q block).
// Q/K tiles staged via Tensor Data Mover; V staged transposed by VALU.
// scores -> softmax (+biases) -> P*V, gated, stored bf16. LDS ~100 KB.
// ---------------------------------------------------------------------------
__global__ __launch_bounds__(256) void tri_attn_kernel(
    const __bf16* __restrict__ qbuf, const __bf16* __restrict__ kbuf,
    const __bf16* __restrict__ vbuf, const float* __restrict__ gbuf,
    const float* __restrict__ trib, const float* __restrict__ maskb,
    __bf16* __restrict__ obuf)
{
  __shared__ __bf16 sQ[QBLK * CH];      //  4 KB  (row-major [64][32])
  __shared__ __bf16 sK[NRES * CH];      // 16 KB  (row-major [256][32])
  __shared__ __bf16 sVt[CH * NRES];     // 16 KB  (transposed [32][256])
  __shared__ float  sS[QBLK * NRES];    // 64 KB

  const int tid  = threadIdx.x;
  const int lane = tid & 31;
  const int wave = tid >> 5;
  const int half = lane >> 4;
  const int lm   = lane & 15;

  const int n  = blockIdx.x;   // 0..255
  const int h  = blockIdx.y;   // 0..3
  const int qb = blockIdx.z;   // 0..3

  const size_t head_off = (size_t)(n * NH + h) * NRES * CH;
  const __bf16* qh = qbuf + head_off + (size_t)qb * QBLK * CH;
  const __bf16* kh = kbuf + head_off;
  const __bf16* vh = vbuf + head_off;
  const float*  tb_base = trib + ((size_t)h * NRES + qb * QBLK) * NRES;

  // Warm the bias panel (64x256 f32) into cache while we stage tiles.
  __builtin_prefetch(tb_base + tid * 64, 0, 1);

#if defined(USE_TDM)
  if (tid < 32) {  // wave 0 issues the DMA descriptors (EXEC-independent)
    tdm_load_1d(qh, sQ, QBLK * CH);
    tdm_load_1d(kh, sK, NRES * CH);
  }
#else
  for (int i = tid * 8; i < QBLK * CH; i += 2048)
    *(v8bf*)&sQ[i] = *(const v8bf*)&qh[i];
  for (int i = tid * 8; i < NRES * CH; i += 2048)
    *(v8bf*)&sK[i] = *(const v8bf*)&kh[i];
#endif

  // V staged transposed so PV B-fragments are contiguous (all waves).
  for (int i = tid * 8; i < NRES * CH; i += 2048) {
    v8bf vv = *(const v8bf*)&vh[i];
    const int r = i >> 5, c = i & 31;
#pragma unroll
    for (int j = 0; j < 8; ++j) sVt[(c + j) * NRES + r] = vv[j];
  }

#if defined(USE_TDM)
  if (tid < 32) __builtin_amdgcn_s_wait_tensorcnt((short)0);
#endif
  __syncthreads();

  // ---- scores: S(64x256) = Q(64x32) * K^T(32x256); one wmma per tile (K=32)
  {
    const int m0 = (wave >> 1) * 16;
    const v16bf aQ = load_A(&sQ[(m0 + lm) * CH + (half << 4)]);
#pragma unroll 1
    for (int t = 0; t < 8; ++t) {
      const int n0 = (((wave & 1) << 3) + t) * 16;
      // B[kk][nn] = k[n0+nn][kk]; element e: K = 16*half+e (contiguous run)
      v16bf b = *(const v16bf*)&sK[(n0 + lm) * CH + (half << 4)];
      v8f acc = {};
      acc = wmma_bf16(aQ, b, acc);
#pragma unroll
      for (int r = 0; r < 8; ++r)
        sS[(m0 + r + 8 * half) * NRES + n0 + lm] = acc[r];
    }
  }
  __syncthreads();

  // ---- softmax with biases; one thread per query row, float4 sweeps
  if (tid < QBLK) {
    float4* row4 = (float4*)&sS[tid * NRES];
    const float4* tb4 = (const float4*)(tb_base + (size_t)tid * NRES);
    const float4* mb4 = (const float4*)(maskb + (size_t)n * NRES);
    float mx = -3.0e38f;
    for (int k = 0; k < NRES / 4; ++k) {
      float4 s = row4[k];
      const float4 t = tb4[k];
      const float4 m = mb4[k];
      s.x += t.x + m.x; s.y += t.y + m.y; s.z += t.z + m.z; s.w += t.w + m.w;
      row4[k] = s;
      mx = fmaxf(mx, fmaxf(fmaxf(s.x, s.y), fmaxf(s.z, s.w)));
    }
    float sum = 0.0f;
    for (int k = 0; k < NRES / 4; ++k) {
      float4 s = row4[k];
      s.x = __expf(s.x - mx); s.y = __expf(s.y - mx);
      s.z = __expf(s.z - mx); s.w = __expf(s.w - mx);
      row4[k] = s;
      sum += s.x + s.y + s.z + s.w;
    }
    const float inv = fast_rcp(sum);
    for (int k = 0; k < NRES / 4; ++k) {
      float4 s = row4[k];
      s.x *= inv; s.y *= inv; s.z *= inv; s.w *= inv;
      row4[k] = s;
    }
  }
  __syncthreads();

  // ---- o(64x32) = P(64x256) * V(256x32); wave -> one 16x16 output tile
  {
    const int m0  = (wave >> 1) * 16;
    const int n0v = (wave & 1) * 16;
    v8f acc = {};
#pragma unroll 1
    for (int kt = 0; kt < 8; ++kt) {
      const float* ap = &sS[(m0 + lm) * NRES + kt * 32 + (half << 4)];
      const v8f f0 = *(const v8f*)(ap);
      const v8f f1 = *(const v8f*)(ap + 8);
      v16bf a;
#pragma unroll
      for (int e = 0; e < 8; ++e) {
        a[e]     = (__bf16)f0[e];
        a[e + 8] = (__bf16)f1[e];
      }
      v16bf b = *(const v16bf*)&sVt[(n0v + lm) * NRES + kt * 32 + (half << 4)];
      acc = wmma_bf16(a, b, acc);
    }
#pragma unroll
    for (int r = 0; r < 8; ++r) {
      const int qrow = qb * QBLK + m0 + r + 8 * half;
      const int j = h * CH + n0v + lm;                  // 0..127
      const size_t gi = ((size_t)n * NRES + qrow) * CIN + j;
      obuf[gi] = (__bf16)(acc[r] * gbuf[gi]);
    }
  }
}

// ---------------------------------------------------------------------------
// Kernel 3: out = o(bf16)[65536,128] @ Wo[128,128] -> fp32
// ---------------------------------------------------------------------------
__global__ __launch_bounds__(256) void tri_out_kernel(
    const __bf16* __restrict__ obuf, const float* __restrict__ Wo,
    float* __restrict__ out)
{
  __shared__ __bf16 sO[128 * CIN];    // 32 KB
  __shared__ __bf16 sWoT[CIN * CIN];  // 32 KB transposed [n][k]

  const int tid  = threadIdx.x;
  const int lane = tid & 31;
  const int wave = tid >> 5;
  const int half = lane >> 4;
  const int lm   = lane & 15;
  const size_t m_base = (size_t)blockIdx.x * 128;

  for (int i = tid * 8; i < 128 * CIN; i += 2048)
    *(v8bf*)&sO[i] = *(const v8bf*)&obuf[m_base * CIN + i];
  for (int i = tid * 4; i < CIN * CIN; i += 1024) {
    const int k = i >> 7, n = i & 127;
    float4 wv = *(const float4*)&Wo[i];
    sWoT[(n + 0) * CIN + k] = (__bf16)wv.x;
    sWoT[(n + 1) * CIN + k] = (__bf16)wv.y;
    sWoT[(n + 2) * CIN + k] = (__bf16)wv.z;
    sWoT[(n + 3) * CIN + k] = (__bf16)wv.w;
  }
  __syncthreads();

  const int m0 = wave * 16;
  v16bf a[4];
#pragma unroll
  for (int kt = 0; kt < 4; ++kt)
    a[kt] = load_A(&sO[(m0 + lm) * CIN + kt * 32 + (half << 4)]);

#pragma unroll 1
  for (int nt = 0; nt < 8; ++nt) {
    const int n0 = nt * 16;
    v8f acc = {};
#pragma unroll
    for (int kt = 0; kt < 4; ++kt) {
      v16bf b = *(const v16bf*)&sWoT[(n0 + lm) * CIN + kt * 32 + (half << 4)];
      acc = wmma_bf16(a[kt], b, acc);
    }
#pragma unroll
    for (int r = 0; r < 8; ++r)
      out[(m_base + m0 + r + 8 * half) * CIN + n0 + lm] = acc[r];
  }
}

// ---------------------------------------------------------------------------
extern "C" void kernel_launch(void* const* d_in, const int* in_sizes, int n_in,
                              void* d_out, int out_size, void* d_ws, size_t ws_size,
                              hipStream_t stream) {
  const float* qx    = (const float*)d_in[0];
  const float* kvx   = (const float*)d_in[1];
  const float* trib  = (const float*)d_in[2];
  const float* maskb = (const float*)d_in[3];
  // d_in[4] = mask (unused; mask_bias carries it in the reference path)
  const float* Wq = (const float*)d_in[5];
  const float* Wk = (const float*)d_in[6];
  const float* Wv = (const float*)d_in[7];
  const float* Wg = (const float*)d_in[8];
  const float* Wo = (const float*)d_in[9];

  // workspace layout (bytes):
  //   q bf16 [N][H][256][32] : 16 MB @ 0
  //   k bf16                 : 16 MB @ 16 MB
  //   v bf16                 : 16 MB @ 32 MB
  //   g fp32 [65536][128]    : 32 MB @ 48 MB
  //   o bf16 [65536][128]    : 16 MB @ 80 MB
  char* ws = (char*)d_ws;
  __bf16* qb = (__bf16*)(ws);
  __bf16* kb = (__bf16*)(ws + (size_t)16 * 1024 * 1024);
  __bf16* vb = (__bf16*)(ws + (size_t)32 * 1024 * 1024);
  float*  gb = (float*)(ws + (size_t)48 * 1024 * 1024);
  __bf16* ob = (__bf16*)(ws + (size_t)80 * 1024 * 1024);

  tri_proj_kernel<<<512, 256, 0, stream>>>(qx, kvx, Wq, Wk, Wv, Wg, qb, kb, vb, gb);

  dim3 agrid(NRES, NH, NRES / QBLK);   // 256 x 4 x 4
  tri_attn_kernel<<<agrid, 256, 0, stream>>>(qb, kb, vb, gb, trib, maskb, ob);

  tri_out_kernel<<<512, 256, 0, stream>>>(ob, Wo, (float*)d_out);
}
